// LocalRPEMultiHeadAttention_53214644797679
// MI455X (gfx1250) — compile-verified, matrix operand order checked by reading, not run on hardware
//
#include <hip/hip_runtime.h>
#include <hip/hip_bf16.h>
#include <stdint.h>

// Problem constants (fixed by the reference).
#define N_NODES  40000
#define M_GROUPS 40000
#define KNBR     16
#define CDIM     128
#define HEADS    4
#define HDIM     32          // CDIM / HEADS
#define BM       4           // groups per block in the fused attention kernel
#define WELEMS   (CDIM * CDIM)   // 16384 elements per weight matrix

typedef __attribute__((ext_vector_type(16))) __bf16 v16bf;
typedef __attribute__((ext_vector_type(8)))  __bf16 v8bf;
typedef __attribute__((ext_vector_type(8)))  float  v8f;
typedef __attribute__((ext_vector_type(4)))  float  v4f;   // builtin vector (not HIP class)

union BF16x16 { v16bf v; v8bf h[2]; };

// Force global (as1) encodings instead of flat_* for hot global traffic.
// Only used with trivial/builtin-vector types (not HIP_vector_type classes).
#define GLD(T, p) (*(const __attribute__((address_space(1))) T*)(unsigned long long)(p))
#define GST(T, p) (*(__attribute__((address_space(1))) T*)(unsigned long long)(p))

#if defined(__gfx1250__)
#define USE_ASYNC_LDS 1
#else
#define USE_ASYNC_LDS 0
#endif

// Async global->LDS copy of 16 bytes (per lane), tracked by ASYNCcnt.
// Inline asm avoids the builtin's CUDA-language-address-space pointer types.
__device__ inline void async_ld16(unsigned lds_addr, const void* gbase, unsigned goff) {
#if USE_ASYNC_LDS
    asm volatile("global_load_async_to_lds_b128 %0, %1, %2"
                 :: "v"(lds_addr), "v"(goff), "s"(gbase)
                 : "memory");
#else
    (void)lds_addr; (void)gbase; (void)goff;
#endif
}

__device__ inline void wait_async0() {
#if USE_ASYNC_LDS
#if __has_builtin(__builtin_amdgcn_s_wait_asynccnt)
    __builtin_amdgcn_s_wait_asynccnt(0);
#else
    asm volatile("s_wait_asynccnt 0x0" ::: "memory");
#endif
#endif
}

__device__ inline void cvt4(const v4f a, __bf16* d) {
    d[0] = (__bf16)a.x; d[1] = (__bf16)a.y; d[2] = (__bf16)a.z; d[3] = (__bf16)a.w;
}

// ---------------------------------------------------------------------------
// Kernel 0: one-time conversion of all five 128x128 weight matrices to bf16.
// 81920 elements, 80 blocks x 256 threads x 4 elements.
// ---------------------------------------------------------------------------
__global__ __launch_bounds__(256)
void wcvt_kernel(const float* __restrict__ Wq, const float* __restrict__ Wk,
                 const float* __restrict__ Wv, const float* __restrict__ Wp,
                 const float* __restrict__ Wvp, __bf16* __restrict__ out)
{
    const int idx = (blockIdx.x * 256 + threadIdx.x) * 4;
    const int mat = idx >> 14;           // 16384 elements per matrix
    const int off = idx & (WELEMS - 1);
    const float* src = (mat == 0) ? Wq : (mat == 1) ? Wk : (mat == 2) ? Wv
                     : (mat == 3) ? Wp : Wvp;
    v4f a = GLD(v4f, src + off);
    __bf16* d = out + ((size_t)mat << 14) + off;
    cvt4(a, d);
}

// ---------------------------------------------------------------------------
// Kernel 1: q/k/v = x @ W^T + b   (N x 128) for three weight sets.
// One block = 16 rows of x; 8 waves; wave w owns output columns [16w, 16w+16).
// A fragments come from a shared bf16 LDS tile (converted once, reused by all
// 8 waves). B fragments are read directly from the pre-converted bf16 weights
// in global memory (each element used exactly once per wave -> no LDS value).
// ---------------------------------------------------------------------------
__global__ __launch_bounds__(256)
void qkv_proj_kernel(const float*  __restrict__ x,
                     const __bf16* __restrict__ wq, const float* __restrict__ bq,
                     const __bf16* __restrict__ wk, const float* __restrict__ bk,
                     const __bf16* __restrict__ wv, const float* __restrict__ bv,
                     float* __restrict__ q_ws, float* __restrict__ k_ws,
                     float* __restrict__ v_ws)
{
    __shared__ __align__(16) __bf16 sIn[16 * CDIM];        // 4 KB

    const int tid  = threadIdx.x;
    const int wave = tid >> 5;
    const int lane = tid & 31;
    const int row0 = blockIdx.x * 16;

    // Stage 16x128 input tile -> bf16 (fully coalesced 8 KB read).
    {
        const float* g = x + (size_t)row0 * CDIM;
        cvt4(GLD(v4f, g + tid * 8 + 0), &sIn[tid * 8 + 0]);
        cvt4(GLD(v4f, g + tid * 8 + 4), &sIn[tid * 8 + 4]);
    }
    __syncthreads();

    const __bf16* Ws[3] = {wq, wk, wv};
    const float*  Bs[3] = {bq, bk, bv};
    float*        Os[3] = {q_ws, k_ws, v_ws};

    const int col0 = wave * 16;
    const int mrow = lane & 15;
    const int hi   = lane >> 4;

    #pragma unroll
    for (int s = 0; s < 3; ++s) {
        const __bf16* W = Ws[s];
        v8f acc = {};
        #pragma unroll
        for (int k0 = 0; k0 < CDIM; k0 += 32) {
            const int kb = k0 + 8 * hi;
            BF16x16 a, b;
            a.h[0] = *reinterpret_cast<const v8bf*>(&sIn[mrow * CDIM + kb]);
            a.h[1] = *reinterpret_cast<const v8bf*>(&sIn[mrow * CDIM + kb + 16]);
            b.h[0] = GLD(v8bf, &W[(col0 + mrow) * CDIM + kb]);
            b.h[1] = GLD(v8bf, &W[(col0 + mrow) * CDIM + kb + 16]);
            acc = __builtin_amdgcn_wmma_f32_16x16x32_bf16(
                      false, a.v, false, b.v, (short)0, acc, false, false);
        }
        const float bias = Bs[s][col0 + mrow];
        float* out = Os[s];
        #pragma unroll
        for (int r = 0; r < 8; ++r) {
            const int row = row0 + r + 8 * hi;
            GST(float, &out[(size_t)row * CDIM + col0 + mrow]) = acc[r] + bias;
        }
    }
}

// ---------------------------------------------------------------------------
// Kernel 2: fused  p/vp projection + gather + scores + softmax + hidden.
// One block = BM(=4) groups -> 64 embed rows. 8 waves.
//   Staging: embed tile converted fp32->bf16; bf16 weights copied to LDS via
//   the CDNA5 async global->LDS path (ASYNCcnt, global_load_async_to_lds_b128).
//   GEMM phase: wave w -> matrix (w&1 : p/vp), row-tile (w>>1), 8 col tiles;
//   p/vp live only in LDS (never hit HBM).
//   Attention phase: 16 (group, head) pairs, 2 per wave.
// Dynamic LDS = 144 KB.
// ---------------------------------------------------------------------------
__global__ __launch_bounds__(256)
void rpe_attn_kernel(const float*  __restrict__ embed,      // (M,K,C)
                     const int*    __restrict__ node_idx,   // (M)
                     const int*    __restrict__ group_idx,  // (M,K)
                     const __bf16* __restrict__ wp_bf, const float* __restrict__ bp,
                     const __bf16* __restrict__ wvp_bf, const float* __restrict__ bvp,
                     const float* __restrict__ q_ws, const float* __restrict__ k_ws,
                     const float* __restrict__ v_ws,
                     float* __restrict__ hidden_out,        // (M,C)
                     float* __restrict__ attn_out)          // (M,H,1,K)
{
    extern __shared__ __align__(16) char smem[];
    __bf16* sE   = reinterpret_cast<__bf16*>(smem);                 // 64*128 bf16 = 16 KB
    __bf16* sWp  = reinterpret_cast<__bf16*>(smem + 16 * 1024);     // 32 KB
    __bf16* sWvp = reinterpret_cast<__bf16*>(smem + 48 * 1024);     // 32 KB
    float*  sP   = reinterpret_cast<float*>(smem + 80 * 1024);      // 32 KB
    float*  sVP  = reinterpret_cast<float*>(smem + 112 * 1024);     // 32 KB

    const int tid  = threadIdx.x;
    const int wave = tid >> 5;
    const int lane = tid & 31;
    const int m0   = blockIdx.x * BM;

    // Stage 64x128 embed tile -> bf16 (coalesced 32 KB read + convert).
    {
        const float* g = embed + (size_t)m0 * KNBR * CDIM;
        #pragma unroll
        for (int i = 0; i < 8; ++i)
            cvt4(GLD(v4f, g + (tid + i * 256) * 4), &sE[(tid + i * 256) * 4]);
    }
    // Stage both pre-converted bf16 weight matrices (pure copy -> async path).
    {
#if USE_ASYNC_LDS
        const unsigned lds_wp  = (unsigned)(unsigned long long)(void*)sWp;
        const unsigned lds_wvp = (unsigned)(unsigned long long)(void*)sWvp;
        #pragma unroll
        for (int i = 0; i < 8; ++i) {
            const unsigned boff = (unsigned)(tid + i * 256) * 16u;  // bytes
            async_ld16(lds_wp  + boff, wp_bf,  boff);
            async_ld16(lds_wvp + boff, wvp_bf, boff);
        }
#else
        #pragma unroll
        for (int i = 0; i < 8; ++i) {
            const int e = (tid + i * 256) * 8;
            *reinterpret_cast<v8bf*>(&sWp[e])  = GLD(v8bf, wp_bf  + e);
            *reinterpret_cast<v8bf*>(&sWvp[e]) = GLD(v8bf, wvp_bf + e);
        }
#endif
    }
    wait_async0();
    __syncthreads();

    // ---- GEMM phase: p = E @ Wp^T + bp ; vp = E @ Wvp^T + bvp  (64 x 128) ----
    const int sel = wave & 1;
    const int rt  = wave >> 1;                 // row tile 0..3
    const __bf16* sWsel = sel ? sWvp : sWp;
    const float*  bsel  = sel ? bvp  : bp;
    float*        osel  = sel ? sVP  : sP;
    const int mrow = lane & 15;
    const int hi   = lane >> 4;

    #pragma unroll
    for (int ct = 0; ct < 8; ++ct) {
        v8f acc = {};
        #pragma unroll
        for (int k0 = 0; k0 < CDIM; k0 += 32) {
            const int kb = k0 + 8 * hi;
            BF16x16 a, b;
            a.h[0] = *reinterpret_cast<const v8bf*>(&sE[(rt * 16 + mrow) * CDIM + kb]);
            a.h[1] = *reinterpret_cast<const v8bf*>(&sE[(rt * 16 + mrow) * CDIM + kb + 16]);
            b.h[0] = *reinterpret_cast<const v8bf*>(&sWsel[(ct * 16 + mrow) * CDIM + kb]);
            b.h[1] = *reinterpret_cast<const v8bf*>(&sWsel[(ct * 16 + mrow) * CDIM + kb + 16]);
            acc = __builtin_amdgcn_wmma_f32_16x16x32_bf16(
                      false, a.v, false, b.v, (short)0, acc, false, false);
        }
        const float bias = bsel[ct * 16 + mrow];
        #pragma unroll
        for (int r = 0; r < 8; ++r)
            osel[(rt * 16 + r + 8 * hi) * CDIM + ct * 16 + mrow] = acc[r] + bias;
    }
    __syncthreads();

    // ---- Attention phase: 16 (m_local, head) pairs; wave handles 2 ----
    const float rscale = 0.17677669529663687f;   // 1/sqrt(32)
    #pragma unroll
    for (int pp = 0; pp < 2; ++pp) {
        const int pair = wave + pp * 8;
        const int ml = pair >> 2;
        const int h  = pair & 3;
        const int m  = m0 + ml;
        const int nidx = node_idx[m];
        const int half = lane >> 4;
        const int j    = lane & 15;
        const float* qh = q_ws + (size_t)nidx * CDIM + h * HDIM + half * 16;

        // scores: lane j handles neighbor j, each half sums 16 channels
        const int gi = group_idx[m * KNBR + j];
        const float* krow = k_ws + (size_t)gi * CDIM + h * HDIM + half * 16;
        const float* prow = &sP[(ml * KNBR + j) * CDIM + h * HDIM + half * 16];
        float acc = 0.f;
        #pragma unroll
        for (int cc = 0; cc < 16; cc += 4) {
            v4f qv = GLD(v4f, qh + cc);
            v4f kv = GLD(v4f, krow + cc);
            v4f pv = *reinterpret_cast<const v4f*>(prow + cc);
            acc += qv.x * (kv.x + pv.x) + qv.y * (kv.y + pv.y)
                 + qv.z * (kv.z + pv.z) + qv.w * (kv.w + pv.w);
        }
        acc += __shfl_xor(acc, 16, 32);          // both halves hold the full dot
        const float score = acc * rscale;

        // softmax over the 16 j-lanes (halves are identical)
        float mx = score;
        mx = fmaxf(mx, __shfl_xor(mx, 1, 32));
        mx = fmaxf(mx, __shfl_xor(mx, 2, 32));
        mx = fmaxf(mx, __shfl_xor(mx, 4, 32));
        mx = fmaxf(mx, __shfl_xor(mx, 8, 32));
        const float e = __expf(score - mx);
        float ssum = e;
        ssum += __shfl_xor(ssum, 1, 32);
        ssum += __shfl_xor(ssum, 2, 32);
        ssum += __shfl_xor(ssum, 4, 32);
        ssum += __shfl_xor(ssum, 8, 32);
        const float at = e / ssum;
        if (half == 0)
            GST(float, &attn_out[((size_t)m * HEADS + h) * KNBR + j]) = at;

        // hidden: 32 lanes = 32 channels of this head
        float hid = 0.f;
        #pragma unroll
        for (int jj = 0; jj < KNBR; ++jj) {
            const float aj  = __shfl(at, jj, 32);
            const int   gij = group_idx[m * KNBR + jj];   // uniform -> scalar load
            hid += aj * (GLD(float, &v_ws[(size_t)gij * CDIM + h * HDIM + lane]) +
                         sVP[(ml * KNBR + jj) * CDIM + h * HDIM + lane]);
        }
        GST(float, &hidden_out[(size_t)m * CDIM + h * HDIM + lane]) = hid;
    }
}

// ---------------------------------------------------------------------------
extern "C" void kernel_launch(void* const* d_in, const int* in_sizes, int n_in,
                              void* d_out, int out_size, void* d_ws, size_t ws_size,
                              hipStream_t stream)
{
    (void)in_sizes; (void)n_in; (void)out_size; (void)ws_size;

    const float* x    = (const float*)d_in[0];
    const float* eqk  = (const float*)d_in[1];
    const int*   nidx = (const int*)  d_in[2];
    const int*   gidx = (const int*)  d_in[3];
    const float* Wq   = (const float*)d_in[4];  const float* bq  = (const float*)d_in[5];
    const float* Wk   = (const float*)d_in[6];  const float* bk  = (const float*)d_in[7];
    const float* Wv   = (const float*)d_in[8];  const float* bv  = (const float*)d_in[9];
    const float* Wp   = (const float*)d_in[10]; const float* bp  = (const float*)d_in[11];
    const float* Wvp  = (const float*)d_in[12]; const float* bvp = (const float*)d_in[13];

    // Workspace: q,k,v (3 * 40000 * 128 fp32 = 61.4 MB, L2-resident for the
    // phase-2 gathers) + 5 bf16 weight matrices (160 KB).
    const size_t NC = (size_t)N_NODES * CDIM;
    float*  q_ws = (float*)d_ws;
    float*  k_ws = q_ws + NC;
    float*  v_ws = k_ws + NC;
    __bf16* wbf  = (__bf16*)(v_ws + NC);
    __bf16* wq_bf  = wbf + 0 * (size_t)WELEMS;
    __bf16* wk_bf  = wbf + 1 * (size_t)WELEMS;
    __bf16* wv_bf  = wbf + 2 * (size_t)WELEMS;
    __bf16* wp_bf  = wbf + 3 * (size_t)WELEMS;
    __bf16* wvp_bf = wbf + 4 * (size_t)WELEMS;

    float* hidden_out = (float*)d_out;                        // (M, C)
    float* attn_out   = hidden_out + (size_t)M_GROUPS * CDIM; // (M, H, 1, K)

    wcvt_kernel<<<80, 256, 0, stream>>>(Wq, Wk, Wv, Wp, Wvp, wbf);

    qkv_proj_kernel<<<N_NODES / 16, 256, 0, stream>>>(
        x, wq_bf, bq, wk_bf, bk, wv_bf, bv, q_ws, k_ws, v_ws);

    rpe_attn_kernel<<<M_GROUPS / BM, 256, 144 * 1024, stream>>>(
        eqk, nidx, gidx, wp_bf, bp, wvp_bf, bvp, q_ws, k_ws, v_ws,
        hidden_out, attn_out);
}
